// PNALayer_35081292874189
// MI455X (gfx1250) — matrix operand order repaired and checked by reading, not compiled, not run on hardware
//
#include <hip/hip_runtime.h>
#include <hip/hip_bf16.h>
#include <math.h>

// ---------------------------------------------------------------------------
// PNA layer, fused for gfx1250 (MI455X): wave32 + v_wmma_f32_16x16x32_bf16
// ---------------------------------------------------------------------------

typedef __attribute__((ext_vector_type(16))) __bf16 v16bf;
typedef __attribute__((ext_vector_type(8)))  float  v8f;

struct __attribute__((aligned(16))) U128 { unsigned x, y, z, w; };
struct __attribute__((aligned(8)))  U64  { unsigned lo, hi; };
union Frag { U128 q[2]; v16bf v; };

__device__ __forceinline__ unsigned short f2bf_u(float f) {
  union { float f; unsigned u; } c; c.f = f;
  unsigned u = c.u;
  u += 0x7FFFu + ((u >> 16) & 1u);      // round-to-nearest-even
  return (unsigned short)(u >> 16);
}

constexpr int B_  = 4;
constexpr int NC  = 512;
constexpr int NF  = 256;

// workspace byte offsets (all 256-aligned); total ~6.4 MB
constexpr size_t OFF_UC   = 0;         // u_c   : B*NC*128 f32   (1 MB)
constexpr size_t OFF_F    = 1048576;   // F     : B*NF*128 f32   (512 KB)
constexpr size_t OFF_W2F  = 1572864;   // U_W2 frags bf16        (32 KB)
constexpr size_t OFF_M1F  = 1605632;   // M_W1 frags bf16        (800 KB)
constexpr size_t OFF_M2F  = 2424832;   // M_W2 frags bf16        (32 KB)
constexpr size_t OFF_AGG  = 2457600;   // agg   : B*NC*4*128 f32 (4 MB)
constexpr size_t OFF_DEGC = 6651904;   // deg_c : B*NC f32
constexpr size_t OFF_DEGF = 6660096;   // deg_f : B*NF f32
constexpr size_t OFF_SCL  = 6664192;   // scale : B*NC f32
constexpr size_t OFF_SCLI = 6672384;   // 1/scale (or 0)

// ---------------------------------------------------------------------------
// Precompute kernels
// ---------------------------------------------------------------------------

// u_c[bc,n] = h_cus[bc,:] @ U_W1[0:64, n] + U_b1[n]
__global__ __launch_bounds__(256) void k_ucus(const float* __restrict__ h_cus,
                                              const float* __restrict__ U_W1,
                                              const float* __restrict__ U_b1,
                                              float* __restrict__ u_c) {
  int idx = blockIdx.x * 256 + threadIdx.x;   // B*NC*128
  int bc = idx >> 7, n = idx & 127;
  const float* hrow = h_cus + bc * 64;
  float s = U_b1[n];
  #pragma unroll 8
  for (int k = 0; k < 64; ++k) s += hrow[k] * U_W1[k * 128 + n];
  u_c[idx] = s;
}

// F[bf,n] = h_fac[bf,:] @ U_W1[64:128, n]
__global__ __launch_bounds__(256) void k_fac(const float* __restrict__ h_fac,
                                             const float* __restrict__ U_W1,
                                             float* __restrict__ Fm) {
  int idx = blockIdx.x * 256 + threadIdx.x;   // B*NF*128
  int bf = idx >> 7, n = idx & 127;
  const float* hrow = h_fac + bf * 64;
  float s = 0.f;
  #pragma unroll 8
  for (int k = 0; k < 64; ++k) s += hrow[k] * U_W1[(64 + k) * 128 + n];
  Fm[idx] = s;
}

// Repack a row-major K x N fp32 weight matrix into bf16 WMMA B-fragments.
// Fragment layout (16x16x32 bf16 B operand): lane = 16*(k_in_tile/16) + n%16,
// element e -> k = kt*32 + 16*(lane/16) + e. Each lane's 16 bf16 contiguous.
__global__ __launch_bounds__(256) void k_repack(const float* __restrict__ W,
                                                unsigned short* __restrict__ dst,
                                                int K, int N) {
  int idx = blockIdx.x * 256 + threadIdx.x;
  if (idx >= K * N) return;
  int k = idx / N, n = idx - k * N;
  int kt = k >> 5, kr = k & 31;
  int lane = ((kr >> 4) << 4) | (n & 15);
  int e  = kr & 15;
  int nt = n >> 4, NT = N >> 4;
  dst[(size_t)((((kt * NT) + nt) << 5) + lane) * 16 + e] = f2bf_u(W[idx]);
}

// degrees: deg_c[b,c] = sum_f adj ; deg_f[b,f] = sum_c adj
__global__ __launch_bounds__(256) void k_deg(const float* __restrict__ adj,
                                             float* __restrict__ deg_c,
                                             float* __restrict__ deg_f) {
  int t = blockIdx.x * 256 + threadIdx.x;
  if (t < B_ * NC) {
    const float* a = adj + (size_t)t * NF;
    float s = 0.f;
    for (int f = 0; f < NF; ++f) s += a[f];
    deg_c[t] = s;
  } else if (t < B_ * NC + B_ * NF) {
    int t2 = t - B_ * NC;
    int b = t2 >> 8, f = t2 & 255;
    const float* a = adj + (size_t)(b * NC) * NF + f;
    float s = 0.f;
    for (int c = 0; c < NC; ++c) s += a[(size_t)c * NF];
    deg_f[t2] = s;
  }
}

// per-batch avg_d, then scale / scale_inv per customer
__global__ __launch_bounds__(256) void k_scale(const float* __restrict__ deg_c,
                                               const float* __restrict__ deg_f,
                                               float* __restrict__ scl,
                                               float* __restrict__ scli) {
  __shared__ float red[256];
  __shared__ float avg_s;
  int b = blockIdx.x, tid = threadIdx.x;
  float s = 0.f;
  for (int i = tid; i < NC + NF; i += 256) {
    float d = (i < NC) ? deg_c[b * NC + i] : deg_f[b * NF + (i - NC)];
    s += logf(d + 1.f);
  }
  red[tid] = s; __syncthreads();
  for (int off = 128; off > 0; off >>= 1) {
    if (tid < off) red[tid] += red[tid + off];
    __syncthreads();
  }
  if (tid == 0) avg_s = red[0] / (float)(NC + NF);
  __syncthreads();
  float avg = avg_s;
  for (int c = tid; c < NC; c += 256) {
    float sc = logf(deg_c[b * NC + c] + 1.f) / avg;
    scl[b * NC + c]  = sc;
    scli[b * NC + c] = (sc != 0.f) ? 1.f / sc : 0.f;
  }
}

// ---------------------------------------------------------------------------
// Fused edge-MLP + masked PNA aggregation. One block per (b,c).
// A = relu(F[b] + u_c[b,c] + e*w129) built in LDS (bf16), then
// y = A @ U_W2 via WMMA; sum/sumsq/max/min accumulated in registers.
// Facilities processed in 2 chunks of 128 rows to keep LDS < 64 KB.
// ---------------------------------------------------------------------------
__global__ __launch_bounds__(256) void k_edge(
    const float* __restrict__ Fmat,            // (B, NF, 128)
    const float* __restrict__ u_c,             // (B*NC, 128)
    const float* __restrict__ U_W1,            // row 128 = edge weights
    const float* __restrict__ U_b2,            // (128)
    const float* __restrict__ edge,            // (B*NC, NF)
    const float* __restrict__ adj,             // (B*NC, NF)
    const unsigned short* __restrict__ W2f,    // U_W2 frags
    float* __restrict__ agg)                   // (B*NC, 4, 128)
{
  extern __shared__ char smem[];
  unsigned short* Asm = (unsigned short*)smem;        // 128 rows x 136 (pad) bf16
  float* u_s   = (float*)(smem + 34816);              // 128
  float* w_s   = (float*)(smem + 35328);              // 128
  float* e_s   = (float*)(smem + 35840);              // 256
  float* adj_s = (float*)(smem + 36864);              // 256  (total 37888 B)

  const int bc  = blockIdx.x;
  const int b   = bc >> 9;
  const int tid = threadIdx.x;

  if (tid < 128) {
    u_s[tid] = u_c[(size_t)bc * 128 + tid];
    w_s[tid] = U_W1[128 * 128 + tid];
  }
  e_s[tid]   = edge[(size_t)bc * NF + tid];
  adj_s[tid] = adj[(size_t)bc * NF + tid];

  // Preload B-fragments of U_W2 (held in VGPRs for the whole kernel)
  const int w = tid >> 5, l = tid & 31;
  const int m = l & 15,  h = l >> 4;
  Frag Bf[4];
  const U128* W2q = (const U128*)W2f;
  #pragma unroll
  for (int kt = 0; kt < 4; ++kt) {
    int base = ((kt * 8 + w) * 32 + l) * 2;
    Bf[kt].q[0] = W2q[base];
    Bf[kt].q[1] = W2q[base + 1];
  }

  float r_sum = 0.f, r_sq = 0.f, r_cnt = 0.f;
  float r_mx = -3.402823466e38f, r_mn = 3.402823466e38f;

  const float4* Fq = (const float4*)(Fmat + (size_t)b * NF * 128);

  for (int part = 0; part < 2; ++part) {
    __syncthreads();   // previous chunk fully consumed / inputs ready

    // phase 1: build 128x128 relu(pre) tile in LDS as bf16
    #pragma unroll 2
    for (int j = 0; j < 16; ++j) {
      int idx = j * 1024 + tid * 4;              // local element in chunk
      float4 fv = Fq[(part * 16384 + idx) >> 2];
      int fl = idx >> 7;                         // local row 0..127
      int k  = idx & 127;
      float e = e_s[part * 128 + fl];
      float p0 = fmaxf(fv.x + u_s[k]     + e * w_s[k],     0.f);
      float p1 = fmaxf(fv.y + u_s[k + 1] + e * w_s[k + 1], 0.f);
      float p2 = fmaxf(fv.z + u_s[k + 2] + e * w_s[k + 2], 0.f);
      float p3 = fmaxf(fv.w + u_s[k + 3] + e * w_s[k + 3], 0.f);
      U64 pk;
      pk.lo = (unsigned)f2bf_u(p0) | ((unsigned)f2bf_u(p1) << 16);
      pk.hi = (unsigned)f2bf_u(p2) | ((unsigned)f2bf_u(p3) << 16);
      *(U64*)(Asm + fl * 136 + k) = pk;
    }
    __syncthreads();

    // phase 2: WMMA GEMM over this chunk + on-the-fly masked aggregation.
    // Wave w owns output N-tile [16w, 16w+16).
    for (int mt = 0; mt < 8; ++mt) {
      v8f acc = {};
      #pragma unroll
      for (int kt = 0; kt < 4; ++kt) {
        Frag a;
        int off = (mt * 16 + m) * 136 + kt * 32 + 16 * h;
        a.q[0] = *(const U128*)(Asm + off);
        a.q[1] = *(const U128*)(Asm + off + 8);
        acc = __builtin_amdgcn_wmma_f32_16x16x32_bf16(
            false, a.v, false, Bf[kt].v, (short)0, acc, false, false);
      }
      // masks for the 8 facility rows this lane-half holds
      const float4* ap = (const float4*)(adj_s + part * 128 + mt * 16 + 8 * h);
      float4 a0 = ap[0], a1 = ap[1];
      float mk[8] = {a0.x, a0.y, a0.z, a0.w, a1.x, a1.y, a1.z, a1.w};
      #pragma unroll
      for (int v = 0; v < 8; ++v) {
        float x = acc[v], mv = mk[v];
        r_sum += mv * x;
        r_sq  += mv * x * x;
        r_cnt += mv;
        r_mx = fmaxf(r_mx, (mv > 0.f) ? x : -3.402823466e38f);
        r_mn = fminf(r_mn, (mv > 0.f) ? x :  3.402823466e38f);
      }
    }
  }

  // combine lane-halves (M 0..7 with M 8..15 per tile)
  r_sum += __shfl_xor(r_sum, 16, 32);
  r_sq  += __shfl_xor(r_sq,  16, 32);
  r_cnt += __shfl_xor(r_cnt, 16, 32);
  r_mx = fmaxf(r_mx, __shfl_xor(r_mx, 16, 32));
  r_mn = fminf(r_mn, __shfl_xor(r_mn, 16, 32));

  if (l < 16) {
    int n = w * 16 + l;
    float b2   = U_b2[n];
    float inv  = 1.f / r_cnt;
    float mean = r_sum * inv;
    float var  = fmaxf(r_sq * inv - mean * mean, 0.f);
    float* ag = agg + (size_t)bc * 512;
    ag[n]       = mean + b2;        // mean  (+U_b2; shift commutes)
    ag[128 + n] = sqrtf(var);       // std   (shift-invariant)
    ag[256 + n] = r_mx + b2;        // max
    ag[384 + n] = r_mn + b2;        // min
  }
}

// ---------------------------------------------------------------------------
// M-MLP: one block per 16 customers. h(1600) built in LDS bf16,
// GEMM1 (K=1600 -> 256) relu, GEMM2 (K=256 -> 64), all via WMMA.
// ---------------------------------------------------------------------------
__global__ __launch_bounds__(256) void k_mmlp(
    const float* __restrict__ h_cus,           // (B*NC, 64)
    const float* __restrict__ agg,             // (B*NC, 512)
    const float* __restrict__ scl,
    const float* __restrict__ scli,
    const unsigned short* __restrict__ M1f,    // M_W1 frags (50 kt x 16 nt)
    const unsigned short* __restrict__ M2f,    // M_W2 frags (8 kt x 4 nt)
    const float* __restrict__ M_b1,
    const float* __restrict__ M_b2,
    float* __restrict__ out)                   // (B*NC, 64)
{
  extern __shared__ char smem[];
  unsigned short* H1 = (unsigned short*)smem;             // 16 x 1608 bf16
  unsigned short* H2 = (unsigned short*)(smem + 51456);   // 16 x 264  bf16
  float* scl_s  = (float*)(smem + 59904);                 // 16
  float* scli_s = scl_s + 16;                             // 16 (total 60032 B)

  const int r0  = blockIdx.x * 16;
  const int tid = threadIdx.x;

  if (tid < 16) {
    scl_s[tid]  = scl[r0 + tid];
    scli_s[tid] = scli[r0 + tid];
  }
  __syncthreads();

  // phase 1: build feature rows (16 x 1600) in bf16
  for (int i = 0; i < 100; ++i) {
    int idx = i * 256 + tid;                   // 16*1600 = 25600
    int r = idx / 1600;
    int j = idx - r * 1600;
    float val;
    if (j < 64) {
      val = h_cus[(size_t)(r0 + r) * 64 + j];
    } else {
      int t2  = j - 64;
      int ia  = t2 / 384;                      // which aggregate
      int rem = t2 - ia * 384;
      int s   = rem >> 7;                      // 0: raw, 1: *scale, 2: *scale_inv
      int k   = rem & 127;
      float base = agg[(size_t)(r0 + r) * 512 + ia * 128 + k];
      float f = (s == 0) ? 1.f : ((s == 1) ? scl_s[r] : scli_s[r]);
      val = base * f;
    }
    H1[r * 1608 + j] = f2bf_u(val);
  }
  __syncthreads();

  const int w = tid >> 5, l = tid & 31;
  const int m = l & 15,  h = l >> 4;

  // phase 2: H2 = relu(H1 @ M_W1 + b1); wave w -> N-tiles {w, w+8}
  {
    v8f acc0 = {}, acc1 = {};
    const int nt0 = w, nt1 = w + 8;
    const U128* M1q = (const U128*)M1f;
    for (int kt = 0; kt < 50; ++kt) {
      Frag a;
      int off = m * 1608 + kt * 32 + 16 * h;
      a.q[0] = *(const U128*)(H1 + off);
      a.q[1] = *(const U128*)(H1 + off + 8);
      Frag b0, b1;
      int base0 = ((kt * 16 + nt0) * 32 + l) * 2;
      int base1 = ((kt * 16 + nt1) * 32 + l) * 2;
      b0.q[0] = M1q[base0]; b0.q[1] = M1q[base0 + 1];
      b1.q[0] = M1q[base1]; b1.q[1] = M1q[base1 + 1];
      acc0 = __builtin_amdgcn_wmma_f32_16x16x32_bf16(
          false, a.v, false, b0.v, (short)0, acc0, false, false);
      acc1 = __builtin_amdgcn_wmma_f32_16x16x32_bf16(
          false, a.v, false, b1.v, (short)0, acc1, false, false);
    }
    #pragma unroll
    for (int t = 0; t < 2; ++t) {
      int n = (t ? nt1 : nt0) * 16 + m;
      float b1v = M_b1[n];
      #pragma unroll
      for (int v = 0; v < 8; ++v) {
        float x = (t ? acc1[v] : acc0[v]) + b1v;
        H2[(8 * h + v) * 264 + n] = f2bf_u(fmaxf(x, 0.f));
      }
    }
  }
  __syncthreads();

  // phase 3: out = H2 @ M_W2 + b2; waves 0..3 -> 4 N-tiles (wave-uniform branch)
  if (w < 4) {
    const U128* M2q = (const U128*)M2f;
    v8f acc = {};
    #pragma unroll
    for (int kt = 0; kt < 8; ++kt) {
      Frag a;
      int off = m * 264 + kt * 32 + 16 * h;
      a.q[0] = *(const U128*)(H2 + off);
      a.q[1] = *(const U128*)(H2 + off + 8);
      Frag bf;
      int base = ((kt * 4 + w) * 32 + l) * 2;
      bf.q[0] = M2q[base]; bf.q[1] = M2q[base + 1];
      acc = __builtin_amdgcn_wmma_f32_16x16x32_bf16(
          false, a.v, false, bf.v, (short)0, acc, false, false);
    }
    int n = w * 16 + m;
    float b2v = M_b2[n];
    #pragma unroll
    for (int v = 0; v < 8; ++v) {
      out[(size_t)(r0 + 8 * h + v) * 64 + n] = acc[v] + b2v;
    }
  }
}

// ---------------------------------------------------------------------------
extern "C" void kernel_launch(void* const* d_in, const int* in_sizes, int n_in,
                              void* d_out, int out_size, void* d_ws, size_t ws_size,
                              hipStream_t stream) {
  (void)in_sizes; (void)n_in; (void)out_size; (void)ws_size;

  const float* h_fac = (const float*)d_in[0];
  const float* h_cus = (const float*)d_in[1];
  const float* edge  = (const float*)d_in[2];
  const float* adj   = (const float*)d_in[3];
  const float* U_W1  = (const float*)d_in[4];
  const float* U_b1  = (const float*)d_in[5];
  const float* U_W2  = (const float*)d_in[6];
  const float* U_b2  = (const float*)d_in[7];
  const float* M_W1  = (const float*)d_in[8];
  const float* M_b1  = (const float*)d_in[9];
  const float* M_W2  = (const float*)d_in[10];
  const float* M_b2  = (const float*)d_in[11];

  char* ws = (char*)d_ws;
  float*          u_c   = (float*)(ws + OFF_UC);
  float*          Fm    = (float*)(ws + OFF_F);
  unsigned short* W2f   = (unsigned short*)(ws + OFF_W2F);
  unsigned short* M1f   = (unsigned short*)(ws + OFF_M1F);
  unsigned short* M2f   = (unsigned short*)(ws + OFF_M2F);
  float*          agg   = (float*)(ws + OFF_AGG);
  float*          deg_c = (float*)(ws + OFF_DEGC);
  float*          deg_f = (float*)(ws + OFF_DEGF);
  float*          scl   = (float*)(ws + OFF_SCL);
  float*          scli  = (float*)(ws + OFF_SCLI);

  k_ucus  <<<1024, 256, 0, stream>>>(h_cus, U_W1, U_b1, u_c);
  k_fac   <<<512,  256, 0, stream>>>(h_fac, U_W1, Fm);
  k_repack<<<64,   256, 0, stream>>>(U_W2, W2f, 128, 128);
  k_repack<<<1600, 256, 0, stream>>>(M_W1, M1f, 1600, 256);
  k_repack<<<64,   256, 0, stream>>>(M_W2, M2f, 256, 64);
  k_deg   <<<12,   256, 0, stream>>>(adj, deg_c, deg_f);
  k_scale <<<4,    256, 0, stream>>>(deg_c, deg_f, scl, scli);

  k_edge<<<B_ * NC, 256, 37888, stream>>>(Fm, u_c, U_W1, U_b2, edge, adj, W2f, agg);
  k_mmlp<<<(B_ * NC) / 16, 256, 60032, stream>>>(h_cus, agg, scl, scli, M1f, M2f,
                                                 M_b1, M_b2, (float*)d_out);
}
